// Seq2Seq_32409823216189
// MI455X (gfx1250) — matrix operand order repaired and checked by reading, not compiled, not run on hardware
//
#include <hip/hip_runtime.h>
#include <math.h>

// ---------------------------------------------------------------------------
// Types for CDNA5 WMMA (wave32): v_wmma_f32_16x16x32_bf16
// ---------------------------------------------------------------------------
typedef __attribute__((ext_vector_type(16))) __bf16 v16bf;
typedef __attribute__((ext_vector_type(8)))  __bf16 v8bf;
typedef __attribute__((ext_vector_type(8)))  float  v8f;

#define Bq 64
#define Sq 512
#define Hq 512
#define Tq 32
#define G4H 2048   // 4*H

#if defined(__has_builtin)
#if __has_builtin(__builtin_amdgcn_sched_barrier)
#define SCHED_FENCE() __builtin_amdgcn_sched_barrier(0)
#endif
#endif
#ifndef SCHED_FENCE
#define SCHED_FENCE() ((void)0)
#endif

// ---------------------------------------------------------------------------
// WMMA helpers
// ---------------------------------------------------------------------------
__device__ __forceinline__ v8f wmma_bf16(v16bf a, v16bf b, v8f c) {
  // (neg_a, A, neg_b, B, c_mod, C, reuse_a, reuse_b)
  return __builtin_amdgcn_wmma_f32_16x16x32_bf16(false, a, false, b, (short)0, c,
                                                 false, false);
}

// A-matrix 16x32 bf16 tile, one row per lane (lanes 0-15: K 0..15;
// lanes 16-31: K 16..31). Two contiguous 8-elem chunks per lane.
__device__ __forceinline__ v16bf load_a_bf16(const __bf16* row, int k0, int khalf) {
  v8bf lo = *reinterpret_cast<const v8bf*>(row + k0 + khalf * 16);
  v8bf hi = *reinterpret_cast<const v8bf*>(row + k0 + 8 + khalf * 16);
  v16bf a;
#pragma unroll
  for (int i = 0; i < 8; ++i) { a[i] = lo[i]; a[i + 8] = hi[i]; }
  return a;
}

// Same A tile but gathered from an fp32 table row (embedding lookup fused).
__device__ __forceinline__ v16bf load_a_f32(const float* row, int k0, int khalf) {
  v8f lo = *reinterpret_cast<const v8f*>(row + k0 + khalf * 16);
  v8f hi = *reinterpret_cast<const v8f*>(row + k0 + 8 + khalf * 16);
  v16bf a;
#pragma unroll
  for (int i = 0; i < 8; ++i) { a[i] = (__bf16)lo[i]; a[i + 8] = (__bf16)hi[i]; }
  return a;
}

// B-matrix 32x16 bf16 tile: lane = column N (lane&15), K-half by lane>>4.
// Source is row-major W[N][K], so 16 contiguous bf16 per lane.
__device__ __forceinline__ v16bf load_b_bf16(const __bf16* p) {
  const v8bf* q = reinterpret_cast<const v8bf*>(p);
  v8bf lo = q[0];
  v8bf hi = q[1];
  v16bf b;
#pragma unroll
  for (int i = 0; i < 8; ++i) { b[i] = lo[i]; b[i + 8] = hi[i]; }
  return b;
}

__device__ __forceinline__ float sigm(float x) { return 1.0f / (1.0f + __expf(-x)); }

// ---------------------------------------------------------------------------
// One fused LSTM time step: z = [tokEmb | x2 | h] @ [Wtok | W2 | Whh]^T + bias
// Each wave owns a 16x16 (batch x hidden-col) tile and computes the i/f/g/o
// gate tiles (4 WMMA accumulators sharing one A tile), then applies the cell
// update in-register. c is fp32 in-place; h written bf16 (host double-buffers).
// Grid: 128 waves = 16 blocks x 256 threads.  All K dims are 512.
//
// Only ~1 wave/SIMD is resident, so all latency hiding is ILP: per operand we
// issue all 4 gate B-tile loads (8 x b128, distinct registers), then a
// sched_barrier pins the WMMA block below the loads so the hardware waits
// stagger (loadcnt 6/4/2/0) instead of serializing load->wmma per gate.
// ---------------------------------------------------------------------------
__global__ __launch_bounds__(256) void lstm_step_kernel(
    const int* __restrict__ tok, int tokStride, const float* __restrict__ table,
    const __bf16* __restrict__ Wtok,
    const __bf16* __restrict__ x2, const __bf16* __restrict__ W2,
    const __bf16* __restrict__ hA, const __bf16* __restrict__ Whh,
    const float* __restrict__ bias,
    float* __restrict__ cst,
    __bf16* __restrict__ h_out,
    __bf16* __restrict__ h_out2, int h2_stride) {
  const int wave  = (blockIdx.x * blockDim.x + threadIdx.x) >> 5;
  const int lane  = threadIdx.x & 31;
  const int mt    = wave >> 5;          // batch tile  0..3
  const int nt    = wave & 31;          // hidden tile 0..31
  const int mrow  = lane & 15;
  const int khalf = lane >> 4;
  const int brow  = mt * 16 + mrow;     // batch row this lane loads for A
  const int ncol  = lane & 15;          // B column within tile
  const int kb    = khalf * 16;         // B K-base within 32-chunk

  v8f acc[4];
#pragma unroll
  for (int g = 0; g < 4; ++g) acc[g] = (v8f)(0.0f);

  const bool hasTok = (tok != nullptr);   // uniform guards -> SALU branches
  const bool hasX2  = (x2 != nullptr);

  const float*  gRow = hasTok ? (table + (long)tok[(long)brow * tokStride] * Hq)
                              : nullptr;
  const __bf16* hRow = hA + (long)brow * Hq;
  const __bf16* xRow = hasX2 ? (x2 + (long)brow * Hq) : nullptr;
  const long    bcolBase = ((long)nt * 16 + ncol) * Hq;

  // Prime the A tiles for k0 = 0.
  v16bf ah = load_a_bf16(hRow, 0, khalf);
  v16bf ax, a2;
  if (hasTok) ax = load_a_f32(gRow, 0, khalf);
  if (hasX2)  a2 = load_a_bf16(xRow, 0, khalf);

  for (int k0 = 0; k0 < Hq; k0 += 32) {
    const long bcol = bcolBase + k0 + kb;

    // Recurrent operand: issue all 4 gate B loads, fence, then 4 WMMAs.
    {
      v16bf b0 = load_b_bf16(Whh + 0L * Hq * Hq + bcol);
      v16bf b1 = load_b_bf16(Whh + 1L * Hq * Hq + bcol);
      v16bf b2 = load_b_bf16(Whh + 2L * Hq * Hq + bcol);
      v16bf b3 = load_b_bf16(Whh + 3L * Hq * Hq + bcol);
      SCHED_FENCE();
      acc[0] = wmma_bf16(ah, b0, acc[0]);
      acc[1] = wmma_bf16(ah, b1, acc[1]);
      acc[2] = wmma_bf16(ah, b2, acc[2]);
      acc[3] = wmma_bf16(ah, b3, acc[3]);
    }
    if (hasTok) {
      v16bf b0 = load_b_bf16(Wtok + 0L * Hq * Hq + bcol);
      v16bf b1 = load_b_bf16(Wtok + 1L * Hq * Hq + bcol);
      v16bf b2 = load_b_bf16(Wtok + 2L * Hq * Hq + bcol);
      v16bf b3 = load_b_bf16(Wtok + 3L * Hq * Hq + bcol);
      SCHED_FENCE();
      acc[0] = wmma_bf16(ax, b0, acc[0]);
      acc[1] = wmma_bf16(ax, b1, acc[1]);
      acc[2] = wmma_bf16(ax, b2, acc[2]);
      acc[3] = wmma_bf16(ax, b3, acc[3]);
    }
    if (hasX2) {
      v16bf b0 = load_b_bf16(W2 + 0L * Hq * Hq + bcol);
      v16bf b1 = load_b_bf16(W2 + 1L * Hq * Hq + bcol);
      v16bf b2 = load_b_bf16(W2 + 2L * Hq * Hq + bcol);
      v16bf b3 = load_b_bf16(W2 + 3L * Hq * Hq + bcol);
      SCHED_FENCE();
      acc[0] = wmma_bf16(a2, b0, acc[0]);
      acc[1] = wmma_bf16(a2, b1, acc[1]);
      acc[2] = wmma_bf16(a2, b2, acc[2]);
      acc[3] = wmma_bf16(a2, b3, acc[3]);
    }

    // Prefetch next iteration's A tiles under the WMMAs above.
    const int kn = k0 + 32;
    if (kn < Hq) {
      ah = load_a_bf16(hRow, kn, khalf);
      if (hasTok) ax = load_a_f32(gRow, kn, khalf);
      if (hasX2)  a2 = load_a_bf16(xRow, kn, khalf);
    }
  }

  // Cell update. C/D layout: VGPR r holds (M = r + 8*khalf, N = lane&15).
  const int j = nt * 16 + ncol;
  const float bi = bias[j];
  const float bf = bias[Hq + j];
  const float bg = bias[2 * Hq + j];
  const float bo = bias[3 * Hq + j];
#pragma unroll
  for (int r = 0; r < 8; ++r) {
    const int  bm  = mt * 16 + r + khalf * 8;
    const long idx = (long)bm * Hq + j;
    float zi = acc[0][r] + bi;
    float zf = acc[1][r] + bf;
    float zg = acc[2][r] + bg;
    float zo = acc[3][r] + bo;
    float cv = cst[idx];
    cv = sigm(zf) * cv + sigm(zi) * tanhf(zg);
    float hv = sigm(zo) * tanhf(cv);
    cst[idx]   = cv;
    h_out[idx] = (__bf16)hv;
    if (h_out2) h_out2[(long)bm * h2_stride + j] = (__bf16)hv;
  }
}

// ---------------------------------------------------------------------------
// Masked softmax attention: attn[b] = softmax(q[b]·K[b,s]/sqrt(H), mask)·K[b]
// One block per batch row. K is bf16 [B, S, H] row-major (b*S+s)*H.
// ---------------------------------------------------------------------------
__global__ __launch_bounds__(256) void attention_kernel(
    const __bf16* __restrict__ q, const __bf16* __restrict__ Kmat,
    const int* __restrict__ src, __bf16* __restrict__ attn, float scale) {
  __shared__ float qs[Hq];
  __shared__ float sc[Sq];
  __shared__ float red[256];
  const int b = blockIdx.x, tid = threadIdx.x;

  for (int j = tid; j < Hq; j += 256) qs[j] = (float)q[b * Hq + j];
  __syncthreads();

  for (int s = tid; s < Sq; s += 256) {
    const __bf16* kr = Kmat + ((long)b * Sq + s) * Hq;
    float a = 0.0f;
    for (int j = 0; j < Hq; ++j) a += qs[j] * (float)kr[j];
    sc[s] = (src[(long)b * Sq + s] != 0) ? a * scale : -1e9f;
  }
  __syncthreads();

  float m = -3.4e38f;
  for (int s = tid; s < Sq; s += 256) m = fmaxf(m, sc[s]);
  red[tid] = m;
  __syncthreads();
  for (int off = 128; off > 0; off >>= 1) {
    if (tid < off) red[tid] = fmaxf(red[tid], red[tid + off]);
    __syncthreads();
  }
  m = red[0];
  __syncthreads();

  float sum = 0.0f;
  for (int s = tid; s < Sq; s += 256) {
    float e = __expf(sc[s] - m);
    sc[s] = e;
    sum += e;
  }
  red[tid] = sum;
  __syncthreads();
  for (int off = 128; off > 0; off >>= 1) {
    if (tid < off) red[tid] += red[tid + off];
    __syncthreads();
  }
  const float inv = 1.0f / red[0];
  __syncthreads();

  for (int j = tid; j < Hq; j += 256) {
    const __bf16* kc = Kmat + (long)b * Sq * Hq + j;
    float a = 0.0f;
    for (int s = 0; s < Sq; ++s) a += sc[s] * (float)kc[(long)s * Hq];
    attn[b * Hq + j] = (__bf16)(a * inv);
  }
}

// ---------------------------------------------------------------------------
// Output projection: out[b, t+1, v] = h1[b]·W_out[:,v] + b_out[v]   (T=32)
// ---------------------------------------------------------------------------
__global__ __launch_bounds__(256) void outproj_kernel(
    const __bf16* __restrict__ h1, const float* __restrict__ W_out,
    const float* __restrict__ b_out, float* __restrict__ out, int t) {
  const int idx = blockIdx.x * 256 + threadIdx.x;  // 0..2047
  const int b = idx >> 5, v = idx & 31;
  float a = b_out[v];
  for (int j = 0; j < Hq; ++j) a += (float)h1[b * Hq + j] * W_out[j * Tq + v];
  out[(long)b * Sq * Tq + (long)(t + 1) * Tq + v] = a;
}

// ---------------------------------------------------------------------------
// fp32 -> bf16 weight conversion (with optional column slicing; cols = 512)
// ---------------------------------------------------------------------------
__global__ __launch_bounds__(256) void cvt_kernel(
    const float* __restrict__ in, __bf16* __restrict__ out,
    int rows, int ld, int coloff) {
  const int i = blockIdx.x * 256 + threadIdx.x;
  if (i >= rows * Hq) return;
  const int r = i >> 9, c = i & 511;
  out[i] = (__bf16)in[(long)r * ld + coloff + c];
}

// ---------------------------------------------------------------------------
extern "C" void kernel_launch(void* const* d_in, const int* in_sizes, int n_in,
                              void* d_out, int out_size, void* d_ws, size_t ws_size,
                              hipStream_t stream) {
  (void)in_sizes; (void)n_in; (void)ws_size;
  const int*   src      = (const int*)d_in[0];
  const int*   trg      = (const int*)d_in[1];
  const float* emb_enc  = (const float*)d_in[2];
  const float* emb_dec  = (const float*)d_in[3];
  const float* eWih0    = (const float*)d_in[4];
  const float* eWhh0    = (const float*)d_in[5];
  const float* eb0      = (const float*)d_in[6];
  const float* eWih1    = (const float*)d_in[7];
  const float* eWhh1    = (const float*)d_in[8];
  const float* eb1      = (const float*)d_in[9];
  const float* dWih0    = (const float*)d_in[10];  // [2048, 1024]
  const float* dWhh0    = (const float*)d_in[11];
  const float* db0      = (const float*)d_in[12];
  const float* dWih1    = (const float*)d_in[13];
  const float* dWhh1    = (const float*)d_in[14];
  const float* db1      = (const float*)d_in[15];
  const float* W_out    = (const float*)d_in[16];
  const float* b_out    = (const float*)d_in[17];
  float* out = (float*)d_out;

  // ---- workspace carve-up (256B aligned) ----
  char* ws = (char*)d_ws;
  size_t off = 0;
  auto alloc = [&](size_t bytes) -> void* {
    void* p = ws + off;
    off = (off + bytes + 255) & ~(size_t)255;
    return p;
  };
  const size_t Wbytes = (size_t)G4H * Hq * sizeof(__bf16);  // 2 MB each
  __bf16* Wih0b  = (__bf16*)alloc(Wbytes);
  __bf16* Whh0b  = (__bf16*)alloc(Wbytes);
  __bf16* Wih1b  = (__bf16*)alloc(Wbytes);
  __bf16* Whh1b  = (__bf16*)alloc(Wbytes);
  __bf16* dWembb = (__bf16*)alloc(Wbytes);   // dec_Wih0[:, :512]
  __bf16* dWab   = (__bf16*)alloc(Wbytes);   // dec_Wih0[:, 512:]
  __bf16* dWhh0b = (__bf16*)alloc(Wbytes);
  __bf16* dWih1b = (__bf16*)alloc(Wbytes);
  __bf16* dWhh1b = (__bf16*)alloc(Wbytes);
  __bf16* Kbf    = (__bf16*)alloc((size_t)Bq * Sq * Hq * sizeof(__bf16));  // 32 MB
  __bf16* h0b[2] = {(__bf16*)alloc(Bq * Hq * 2), (__bf16*)alloc(Bq * Hq * 2)};
  __bf16* h1b[2] = {(__bf16*)alloc(Bq * Hq * 2), (__bf16*)alloc(Bq * Hq * 2)};
  __bf16* attnb  = (__bf16*)alloc(Bq * Hq * 2);
  float*  c0     = (float*)alloc(Bq * Hq * 4);
  float*  c1     = (float*)alloc(Bq * Hq * 4);

  // ---- init ----
  hipMemsetAsync(d_out, 0, (size_t)out_size * sizeof(float), stream);
  hipMemsetAsync(c0, 0, Bq * Hq * 4, stream);
  hipMemsetAsync(c1, 0, Bq * Hq * 4, stream);
  hipMemsetAsync(h0b[0], 0, Bq * Hq * 2, stream);
  hipMemsetAsync(h0b[1], 0, Bq * Hq * 2, stream);
  hipMemsetAsync(h1b[0], 0, Bq * Hq * 2, stream);
  hipMemsetAsync(h1b[1], 0, Bq * Hq * 2, stream);

  // ---- bf16 weight conversion (weights then live in L2: 18 MB << 192 MB) ----
  const int cvtGrid = (G4H * Hq + 255) / 256;
  cvt_kernel<<<cvtGrid, 256, 0, stream>>>(eWih0, Wih0b, G4H, Hq, 0);
  cvt_kernel<<<cvtGrid, 256, 0, stream>>>(eWhh0, Whh0b, G4H, Hq, 0);
  cvt_kernel<<<cvtGrid, 256, 0, stream>>>(eWih1, Wih1b, G4H, Hq, 0);
  cvt_kernel<<<cvtGrid, 256, 0, stream>>>(eWhh1, Whh1b, G4H, Hq, 0);
  cvt_kernel<<<cvtGrid, 256, 0, stream>>>(dWih0, dWembb, G4H, 2 * Hq, 0);
  cvt_kernel<<<cvtGrid, 256, 0, stream>>>(dWih0, dWab,   G4H, 2 * Hq, Hq);
  cvt_kernel<<<cvtGrid, 256, 0, stream>>>(dWhh0, dWhh0b, G4H, Hq, 0);
  cvt_kernel<<<cvtGrid, 256, 0, stream>>>(dWih1, dWih1b, G4H, Hq, 0);
  cvt_kernel<<<cvtGrid, 256, 0, stream>>>(dWhh1, dWhh1b, G4H, Hq, 0);

  const float scale = 0.044194173824159216f;  // 1/sqrt(512)
  int p0 = 0, p1 = 0;

  // ---- encoder: interleave layer-0 and layer-1 steps ----
  for (int s = 0; s < Sq; ++s) {
    lstm_step_kernel<<<16, 256, 0, stream>>>(
        src + s, Sq, emb_enc, Wih0b,          // token-gather operand
        nullptr, nullptr,                      // no dense x2 operand
        h0b[p0], Whh0b, eb0, c0, h0b[1 - p0], nullptr, 0);
    p0 ^= 1;
    lstm_step_kernel<<<16, 256, 0, stream>>>(
        nullptr, 0, nullptr, nullptr,
        h0b[p0], Wih1b,                        // layer-0 h is layer-1 input
        h1b[p1], Whh1b, eb1, c1, h1b[1 - p1],
        Kbf + (long)s * Hq, Sq * Hq);          // also scatter into K matrix
    p1 ^= 1;
  }

  // ---- initial attention from final top-layer encoder hidden ----
  attention_kernel<<<Bq, 256, 0, stream>>>(h1b[p1], Kbf, src, attnb, scale);

  // ---- teacher-forced decoder ----
  for (int t = 0; t < Sq - 1; ++t) {
    lstm_step_kernel<<<16, 256, 0, stream>>>(
        trg + t, Sq, emb_dec, dWembb,          // emb_dec[trg[:,t]] part
        attnb, dWab,                           // attention part of concat input
        h0b[p0], dWhh0b, db0, c0, h0b[1 - p0], nullptr, 0);
    p0 ^= 1;
    lstm_step_kernel<<<16, 256, 0, stream>>>(
        nullptr, 0, nullptr, nullptr,
        h0b[p0], dWih1b,
        h1b[p1], dWhh1b, db1, c1, h1b[1 - p1], nullptr, 0);
    p1 ^= 1;
    outproj_kernel<<<(Bq * Tq) / 256, 256, 0, stream>>>(h1b[p1], W_out, b_out, out, t);
    attention_kernel<<<Bq, 256, 0, stream>>>(h1b[p1], Kbf, src, attnb, scale);
  }
}
